// DIMESDenseEncoder_10617159156313
// MI455X (gfx1250) — matrix operand order repaired and checked by reading, not compiled, not run on hardware
//
#include <hip/hip_runtime.h>
#include <cstdint>

#define BB 16
#define NNODE 200
#define UU 64
#define LL 3
#define EE 199           // N-1 edges per node
#define BN (BB*NNODE)    // 3200
#define EDGE_ROWS (BN*EE) // 636800
#define EPSF 1e-5f

typedef __attribute__((ext_vector_type(16))) __bf16 v16bf;
typedef __attribute__((ext_vector_type(8)))  __bf16 v8bf;
typedef __attribute__((ext_vector_type(4)))  __bf16 v4bf;
typedef __attribute__((ext_vector_type(8)))  float  v8f;

__device__ __forceinline__ float lrelu(float x){ return x > 0.f ? x : 0.01f*x; }

// order-preserving float<->uint encoding for atomic max
__device__ __forceinline__ unsigned fenc(float f){
  unsigned u = __float_as_uint(f);
  return (u & 0x80000000u) ? ~u : (u | 0x80000000u);
}
__device__ __forceinline__ float fdec(unsigned e){
  unsigned u = (e & 0x80000000u) ? (e ^ 0x80000000u) : ~e;
  return __uint_as_float(u);
}

// ---- WMMA helpers (CDNA5 wave32, 16x16x32 bf16 -> f32) ----

__device__ __forceinline__ v8f wmma_bf16(v16bf a, v16bf b, v8f c){
  return __builtin_amdgcn_wmma_f32_16x16x32_bf16(false, a, false, b, (short)0, c, false, false);
}

// A: 16x64 bf16 tile in LDS (row-major [16][64]); gather 16x32 K-chunk kc.
// Per-lane elements are 16 contiguous k: k in [16*half, 16*half+16).
__device__ __forceinline__ v16bf gatherA(const __bf16* sA, int lane, int kc){
  const int M = lane & 15, half = lane >> 4;
  const __bf16* base = sA + M*64 + kc*32 + half*16;
  v8bf lo = *(const v8bf*)(base);        // ds_load_b128
  v8bf hi = *(const v8bf*)(base + 8);    // ds_load_b128
  return __builtin_shufflevector(lo, hi, 0,1,2,3,4,5,6,7,8,9,10,11,12,13,14,15);
}

// B: weight staged TRANSPOSED in LDS as [col][k] (64x64 bf16); per-lane
// fragment is 16 contiguous bf16 -> two ds_load_b128.
__device__ __forceinline__ v16bf gatherBT(const __bf16* sBt, int lane, int kc, int nt){
  const int col = nt*16 + (lane & 15), half = lane >> 4;
  const __bf16* base = sBt + col*64 + kc*32 + half*16;
  v8bf lo = *(const v8bf*)(base);
  v8bf hi = *(const v8bf*)(base + 8);
  return __builtin_shufflevector(lo, hi, 0,1,2,3,4,5,6,7,8,9,10,11,12,13,14,15);
}

// stage a 64x64 f32 weight (row-major [k][col]) into LDS transposed bf16 [col][k]
__device__ __forceinline__ void stageBT(__bf16* sBt, const float* __restrict__ w, int lane){
#pragma unroll 4
  for (int i = lane; i < 1024; i += 32){
    int k = i >> 4, c = (i & 15) * 4;
    float4 v = ((const float4*)w)[i];
    sBt[(c+0)*64 + k] = (__bf16)v.x;
    sBt[(c+1)*64 + k] = (__bf16)v.y;
    sBt[(c+2)*64 + k] = (__bf16)v.z;
    sBt[(c+3)*64 + k] = (__bf16)v.w;
  }
}

// ---- h = lrelu(x @ v_lin0_w + b), x:(B,N,2) ----
__global__ void init_h_kernel(const float* __restrict__ x, const float* __restrict__ w,
                              const float* __restrict__ b, float* __restrict__ h){
  int i = blockIdx.x*blockDim.x + threadIdx.x;
  if (i >= BN*UU) return;
  int bn = i >> 6, u = i & 63;
  h[i] = lrelu(x[bn*2]*w[u] + x[bn*2+1]*w[64+u] + b[u]);
}

// ---- w = lrelu(adj[:,SRC,DST,None]*e_lin0_w + e_lin0_b) ----
__global__ void init_w_kernel(const float* __restrict__ adj, const float* __restrict__ ew0,
                              const float* __restrict__ eb0, float* __restrict__ w){
  size_t i = (size_t)blockIdx.x*blockDim.x + threadIdx.x;   // float4 index
  const size_t total = (size_t)EDGE_ROWS*16;
  if (i >= total) return;
  int row = (int)(i >> 4), c4 = (int)(i & 15);
  int bn = row / EE, m = row % EE;
  int b = bn / NNODE, n = bn % NNODE;
  int dst = m + (m >= n);
  float a = adj[(size_t)b*NNODE*NNODE + (size_t)n*NNODE + dst];
  float4 wv = ((const float4*)ew0)[c4];
  float4 bv = ((const float4*)eb0)[c4];
  float4 o;
  o.x = lrelu(a*wv.x + bv.x); o.y = lrelu(a*wv.y + bv.y);
  o.z = lrelu(a*wv.z + bv.z); o.w = lrelu(a*wv.w + bv.w);
  ((float4*)(w + (size_t)row*UU))[c4] = o;
}

// ---- xk = h @ v_wk + v_bk via WMMA; one single-wave WG per (16-row tile, matrix) ----
__global__ __launch_bounds__(32) void node_gemm_kernel(
    const float* __restrict__ h,
    const float* __restrict__ w1, const float* __restrict__ w2,
    const float* __restrict__ w3, const float* __restrict__ w4,
    const float* __restrict__ b1, const float* __restrict__ b2,
    const float* __restrict__ b3, const float* __restrict__ b4,
    float* __restrict__ xbuf)
{
  __shared__ alignas(16) __bf16 sBt[64*64];
  __shared__ alignas(16) __bf16 sA[16*64];
  const int lane = threadIdx.x;
  const int mat = blockIdx.x & 3;
  const int tileRow = blockIdx.x >> 2;   // 0..199
  const int half = lane >> 4;

  const float* wm = (mat==0) ? w1 : (mat==1) ? w2 : (mat==2) ? w3 : w4;
  const float* bm = (mat==0) ? b1 : (mat==1) ? b2 : (mat==2) ? b3 : b4;
  float* outp = xbuf + (size_t)mat*BN*UU;

  stageBT(sBt, wm, lane);
  const float* hrow0 = h + (size_t)tileRow*16*UU;
#pragma unroll
  for (int i = 0; i < 8; ++i){
    int idx = lane + i*32;
    float4 v = ((const float4*)hrow0)[idx];
    v4bf pk = { (__bf16)v.x, (__bf16)v.y, (__bf16)v.z, (__bf16)v.w };
    *(v4bf*)(sA + idx*4) = pk;           // ds_store_b64
  }
  __syncthreads();

  v8f acc[4];
#pragma unroll
  for (int nt = 0; nt < 4; ++nt)
#pragma unroll
    for (int j = 0; j < 8; ++j) acc[nt][j] = 0.f;

#pragma unroll
  for (int kc = 0; kc < 2; ++kc){
    v16bf a = gatherA(sA, lane, kc);
#pragma unroll
    for (int nt = 0; nt < 4; ++nt)
      acc[nt] = wmma_bf16(a, gatherBT(sBt, lane, kc, nt), acc[nt]);
  }
#pragma unroll
  for (int nt = 0; nt < 4; ++nt){
    int col = nt*16 + (lane & 15);
    float bv = bm[col];
#pragma unroll
    for (int r = 0; r < 8; ++r){
      int row = r + half*8;
      outp[(size_t)(tileRow*16+row)*UU + col] = acc[nt][r] + bv;
    }
  }
}

// ---- edge pass: one single-wave WG per (b,n) node, 199 edge rows in 13 tiles ----
// PASS 1: stats (sum/sumsq per channel) + pooled = max_m sigmoid(w)*x2[DST]
// PASS 2: w += lrelu(bn(w@e_w + e_b + x3 + x4[DST]))  (recompute t; update in place)
template<int PASS>
__global__ __launch_bounds__(32) void edge_pass_kernel(
    const float* __restrict__ w, float* __restrict__ w_out,
    const float* __restrict__ e_w, const float* __restrict__ e_b,
    const float* __restrict__ x2, const float* __restrict__ x3,
    const float* __restrict__ x4,
    float* __restrict__ pooled, float* __restrict__ stats,
    const float* __restrict__ bn_g, const float* __restrict__ bn_b)
{
  __shared__ alignas(16) __bf16 sBt[64*64];  // e_w bf16, transposed [col][k]
  __shared__ alignas(16) __bf16 sA[16*64];   // current row tile bf16
  __shared__ alignas(16) float  sW[16*64];   // current row tile f32
  __shared__ unsigned sPool[64];

  const int lane = threadIdx.x;
  const int node = blockIdx.x;               // b*200+n
  const int b = node / NNODE, n = node % NNODE;
  const int half = lane >> 4;

  stageBT(sBt, e_w, lane);
  if (PASS == 1){ sPool[lane] = fenc(-3.0e38f); sPool[lane+32] = fenc(-3.0e38f); }
  __syncthreads();

  // B fragments held in registers across all tiles
  v16bf bf[8];
#pragma unroll
  for (int kc = 0; kc < 2; ++kc)
#pragma unroll
    for (int nt = 0; nt < 4; ++nt)
      bf[kc*4+nt] = gatherBT(sBt, lane, kc, nt);

  // per-lane column constants (4 cols per lane)
  float ebv[4], x3v[4], gv[4], bbv[4], mv[4], iv[4];
  const float* x3row = x3 + (size_t)node*UU;
#pragma unroll
  for (int nt = 0; nt < 4; ++nt){
    int col = nt*16 + (lane & 15);
    ebv[nt] = e_b[col];
    x3v[nt] = x3row[col];
    if (PASS == 2){
      gv[nt] = bn_g[col]; bbv[nt] = bn_b[col];
      const float cnt = (float)EDGE_ROWS;
      float mean = stats[col] / cnt;
      float var  = stats[64+col]/cnt - mean*mean;
      mv[nt] = mean;
      iv[nt] = rsqrtf(var + EPSF);
    } else { gv[nt]=bbv[nt]=mv[nt]=iv[nt]=0.f; }
  }

  float lsum[4] = {0,0,0,0}, lsq[4] = {0,0,0,0};
  const float* wrow0 = w + (size_t)node*EE*UU;

  // software-pipelined tile loop: prefetch next tile while computing current
  float4 cur[8];
#pragma unroll
  for (int i = 0; i < 8; ++i){
    int idx = lane + i*32;
    int row = idx >> 4, c4 = idx & 15;
    cur[i] = make_float4(0.f,0.f,0.f,0.f);
    if (row < EE) cur[i] = *(const float4*)(wrow0 + (size_t)row*UU + c4*4);
  }

  for (int tile = 0; tile < 13; ++tile){
    // stage current tile into LDS (f32 + packed bf16); fuse pool in PASS 1
#pragma unroll
    for (int i = 0; i < 8; ++i){
      int idx = lane + i*32;               // float4 slot, 256 total
      int row = idx >> 4, c4 = idx & 15;
      int m = tile*16 + row;
      float4 v = cur[i];
      *(float4*)(sW + idx*4) = v;          // ds_store_b128
      v4bf pk = { (__bf16)v.x, (__bf16)v.y, (__bf16)v.z, (__bf16)v.w };
      *(v4bf*)(sA + idx*4) = pk;           // ds_store_b64
      if (PASS == 1 && m < EE){
        int dst = m + (m >= n);
        float4 xv = *(const float4*)(x2 + (size_t)(b*NNODE+dst)*UU + c4*4);
        atomicMax(&sPool[c4*4+0], fenc(xv.x / (1.f + __expf(-v.x))));
        atomicMax(&sPool[c4*4+1], fenc(xv.y / (1.f + __expf(-v.y))));
        atomicMax(&sPool[c4*4+2], fenc(xv.z / (1.f + __expf(-v.z))));
        atomicMax(&sPool[c4*4+3], fenc(xv.w / (1.f + __expf(-v.w))));
      }
    }
    // issue next tile's global loads before the barrier so they overlap WMMA
    float4 nxt[8];
    if (tile < 12){
#pragma unroll
      for (int i = 0; i < 8; ++i){
        int idx = lane + i*32;
        int row = idx >> 4, c4 = idx & 15;
        int m = (tile+1)*16 + row;
        nxt[i] = make_float4(0.f,0.f,0.f,0.f);
        if (m < EE) nxt[i] = *(const float4*)(wrow0 + (size_t)m*UU + c4*4);
      }
    }
    __syncthreads();

    v8f acc[4];
#pragma unroll
    for (int nt = 0; nt < 4; ++nt)
#pragma unroll
      for (int j = 0; j < 8; ++j) acc[nt][j] = 0.f;

#pragma unroll
    for (int kc = 0; kc < 2; ++kc){
      v16bf a = gatherA(sA, lane, kc);
#pragma unroll
      for (int nt = 0; nt < 4; ++nt)
        acc[nt] = wmma_bf16(a, bf[kc*4+nt], acc[nt]);
    }

#pragma unroll
    for (int nt = 0; nt < 4; ++nt){
      int col = nt*16 + (lane & 15);
#pragma unroll
      for (int r = 0; r < 8; ++r){
        int row = r + half*8;
        int m = tile*16 + row;
        if (m < EE){
          int dst = m + (m >= n);
          float t = acc[nt][r] + ebv[nt] + x3v[nt]
                  + x4[(size_t)(b*NNODE+dst)*UU + col];
          if (PASS == 1){
            lsum[nt] += t; lsq[nt] += t*t;
          } else {
            float z = gv[nt]*(t - mv[nt])*iv[nt] + bbv[nt];
            w_out[(size_t)node*EE*UU + (size_t)m*UU + col] = sW[row*64+col] + lrelu(z);
          }
        }
      }
    }
    __syncthreads();
    if (tile < 12){
#pragma unroll
      for (int i = 0; i < 8; ++i) cur[i] = nxt[i];
    }
  }

  if (PASS == 1){
#pragma unroll
    for (int nt = 0; nt < 4; ++nt){
      float s = lsum[nt] + __shfl_down(lsum[nt], 16, 32);
      float q = lsq[nt]  + __shfl_down(lsq[nt],  16, 32);
      if (half == 0){
        int col = nt*16 + lane;
        atomicAdd(&stats[col], s);
        atomicAdd(&stats[64+col], q);
      }
    }
    float* prow = pooled + (size_t)node*UU;
    prow[lane]      = fdec(sPool[lane]);
    prow[lane + 32] = fdec(sPool[lane+32]);
  }
}

// ---- node BN + residual: h += lrelu(bn(x1+pooled)); one WG per channel ----
__global__ __launch_bounds__(256) void node_bn_kernel(
    const float* __restrict__ x1, const float* __restrict__ pooled,
    float* __restrict__ h, const float* __restrict__ g, const float* __restrict__ bb)
{
  __shared__ float sS[256], sQ[256];
  __shared__ float sMean, sInv;
  const int c = blockIdx.x, tid = threadIdx.x;
  float s = 0.f, q = 0.f;
  for (int i = tid; i < BN; i += 256){
    float t = x1[(size_t)i*UU + c] + pooled[(size_t)i*UU + c];
    s += t; q += t*t;
  }
  sS[tid]=s; sQ[tid]=q; __syncthreads();
  for (int off=128; off>0; off>>=1){
    if (tid<off){ sS[tid]+=sS[tid+off]; sQ[tid]+=sQ[tid+off]; }
    __syncthreads();
  }
  if (tid==0){
    float mean = sS[0]/(float)BN;
    float var  = sQ[0]/(float)BN - mean*mean;
    sMean = mean; sInv = rsqrtf(var + EPSF);
  }
  __syncthreads();
  float gg = g[c], bv = bb[c], mean = sMean, inv = sInv;
  for (int i = tid; i < BN; i += 256){
    float t = x1[(size_t)i*UU + c] + pooled[(size_t)i*UU + c];
    h[(size_t)i*UU + c] += lrelu(gg*(t-mean)*inv + bv);
  }
}

// ---- out: one wave per edge row; dot(w_row, e_lin1_w) + b, scatter off-diagonal ----
__global__ __launch_bounds__(256) void out_kernel(
    const float* __restrict__ w, const float* __restrict__ e1w,
    const float* __restrict__ e1b, float* __restrict__ out)
{
  int wid = (int)((blockIdx.x*256u + threadIdx.x) >> 5);
  int lane = threadIdx.x & 31;
  if (wid >= EDGE_ROWS) return;
  int bn = wid / EE, m = wid % EE;
  int b = bn / NNODE, n = bn % NNODE;
  float2 v = ((const float2*)(w + (size_t)wid*UU))[lane];
  float2 e = ((const float2*)e1w)[lane];
  float p = v.x*e.x + v.y*e.y;
#pragma unroll
  for (int off = 16; off > 0; off >>= 1) p += __shfl_down(p, off, 32);
  if (lane == 0){
    int nn = m + (m >= n);
    out[(size_t)b*NNODE*NNODE + (size_t)n*NNODE + nn] = p + e1b[0];
  }
}

extern "C" void kernel_launch(void* const* d_in, const int* in_sizes, int n_in,
                              void* d_out, int out_size, void* d_ws, size_t ws_size,
                              hipStream_t stream)
{
  const float* x        = (const float*)d_in[0];
  const float* adj      = (const float*)d_in[1];
  const float* v_lin0_w = (const float*)d_in[2];
  const float* v_lin0_b = (const float*)d_in[3];
  const float* v_w1 = (const float*)d_in[4];
  const float* v_b1 = (const float*)d_in[5];
  const float* v_w2 = (const float*)d_in[6];
  const float* v_b2 = (const float*)d_in[7];
  const float* v_w3 = (const float*)d_in[8];
  const float* v_b3 = (const float*)d_in[9];
  const float* v_w4 = (const float*)d_in[10];
  const float* v_b4 = (const float*)d_in[11];
  const float* v_bn_g = (const float*)d_in[12];
  const float* v_bn_b = (const float*)d_in[13];
  const float* e_lin0_w = (const float*)d_in[14];
  const float* e_lin0_b = (const float*)d_in[15];
  const float* e_w = (const float*)d_in[16];
  const float* e_b = (const float*)d_in[17];
  const float* e_bn_g = (const float*)d_in[18];
  const float* e_bn_b = (const float*)d_in[19];
  const float* e_lin1_w = (const float*)d_in[20];
  const float* e_lin1_b = (const float*)d_in[21];

  float* out_f  = (float*)d_out;

  float* ws     = (float*)d_ws;
  float* w      = ws;                                  // 40,755,200 f
  float* h      = w + (size_t)EDGE_ROWS*UU;            // 204,800 f
  float* xbuf   = h + (size_t)BN*UU;                   // 4*204,800 f (x1|x2|x3|x4)
  float* pooled = xbuf + (size_t)4*BN*UU;              // 204,800 f
  float* stats  = pooled + (size_t)BN*UU;              // 128 f

  (void)hipMemsetAsync(d_out, 0, (size_t)BB*NNODE*NNODE*sizeof(float), stream);

  init_h_kernel<<<(BN*UU+255)/256, 256, 0, stream>>>(x, v_lin0_w, v_lin0_b, h);
  {
    size_t tot = (size_t)EDGE_ROWS*16;
    init_w_kernel<<<(unsigned)((tot+255)/256), 256, 0, stream>>>(adj, e_lin0_w, e_lin0_b, w);
  }

  for (int layer = 0; layer < LL; ++layer){
    (void)hipMemsetAsync(stats, 0, 128*sizeof(float), stream);
    node_gemm_kernel<<<800, 32, 0, stream>>>(h,
        v_w1 + layer*UU*UU, v_w2 + layer*UU*UU, v_w3 + layer*UU*UU, v_w4 + layer*UU*UU,
        v_b1 + layer*UU, v_b2 + layer*UU, v_b3 + layer*UU, v_b4 + layer*UU, xbuf);
    edge_pass_kernel<1><<<BN, 32, 0, stream>>>(w, w,
        e_w + layer*UU*UU, e_b + layer*UU,
        xbuf + (size_t)1*BN*UU, xbuf + (size_t)2*BN*UU, xbuf + (size_t)3*BN*UU,
        pooled, stats, e_bn_g + layer*UU, e_bn_b + layer*UU);
    node_bn_kernel<<<64, 256, 0, stream>>>(xbuf, pooled, h,
        v_bn_g + layer*UU, v_bn_b + layer*UU);
    edge_pass_kernel<2><<<BN, 32, 0, stream>>>(w, w,
        e_w + layer*UU*UU, e_b + layer*UU,
        xbuf + (size_t)1*BN*UU, xbuf + (size_t)2*BN*UU, xbuf + (size_t)3*BN*UU,
        pooled, stats, e_bn_g + layer*UU, e_bn_b + layer*UU);
  }

  out_kernel<<<(EDGE_ROWS+7)/8, 256, 0, stream>>>(w, e_lin1_w, e_lin1_b, out_f);
}
